// PointNet2Encoder_10531259810452
// MI455X (gfx1250) — matrix operand order repaired and checked by reading, not compiled, not run on hardware
//
#include <hip/hip_runtime.h>
#include <hip/hip_bf16.h>

typedef _Float16 v16h __attribute__((ext_vector_type(16)));
typedef _Float16 v8h  __attribute__((ext_vector_type(8)));
typedef float    v8f  __attribute__((ext_vector_type(8)));

// ---------------------------------------------------------------------------
// Weight pre-swizzle: f32 [Kin x N] row-major -> f16 WMMA B-fragment tile order.
// Tile t = kt*(N/16)+nt (32x16 of K x N). Within tile: lane l, element e:
//   k = kt*32 + (l>>4)*16 + e ; n = nt*16 + (l&15)
// so a B fragment is one contiguous v16h per lane (32B/lane, coalesced).
//
// feat3 != 0 applies a K-row permutation matching the "feats-first" A0 layout:
//   k in [0, feat3)        -> source row 3 + k      (feature weights)
//   k in [feat3, feat3+3)  -> source row k - feat3  (xyz weights)
//   else                   -> zero pad
// ---------------------------------------------------------------------------
__global__ void swz_kernel(const float* __restrict__ W, _Float16* __restrict__ dst,
                           int Kin, int Kpad, int N, int feat3) {
  int tid = blockIdx.x * blockDim.x + threadIdx.x;
  int total = Kpad * N;
  if (tid >= total) return;
  int e = tid & 15;
  int l = (tid >> 4) & 31;
  int t = tid >> 9;
  int tn = N >> 4;
  int kt = t / tn, nt = t - kt * tn;
  int k = kt * 32 + ((l >> 4) << 4) + e;
  int n = nt * 16 + (l & 15);
  int srcK;
  if (feat3 == 0) {
    srcK = k;
  } else {
    if (k < feat3)          srcK = k + 3;
    else if (k < feat3 + 3) srcK = k - feat3;
    else                    srcK = Kin;  // pad
  }
  float v = (srcK < Kin) ? W[(size_t)srcK * N + n] : 0.0f;
  dst[tid] = (_Float16)v;
}

// ---------------------------------------------------------------------------
// WMMA fragment helpers (wave32, 16x16x32 f16 -> f32)
// A (16x32, 16-bit): lane row = lane&15; K base = (lane>>4)*16;
//   halves 0..15 = K kb..kb+15 (matches swizzled B order)
// ---------------------------------------------------------------------------
__device__ __forceinline__ v16h load_a_frag(const _Float16* As, int lda,
                                            int mt, int kt, int lane) {
  int row = mt * 16 + (lane & 15);
  int kb  = kt * 32 + ((lane >> 4) << 4);
  const _Float16* p = As + row * lda + kb;
  v8h lo = *(const v8h*)p;
  v8h hi = *(const v8h*)(p + 8);
  v16h a;
#pragma unroll
  for (int i = 0; i < 8; ++i) { a[i] = lo[i]; a[i + 8] = hi[i]; }
  return a;
}

// Block-level GEMM layer: C = relu(A @ W + bias), A in LDS (M x lda f16),
// W pre-swizzled in global, C written to LDS (f16, row-major, stride ldc).
// Tile loop index is forced scalar (wave id is uniform) so the loop lowers
// as a uniform SALU loop and EXEC is structurally all-1s at every WMMA.
__device__ __forceinline__ void gemm_relu(const _Float16* As, int lda,
                                          const v16h* __restrict__ Bsw,
                                          const float* __restrict__ bias,
                                          _Float16* Cs, int ldc,
                                          int TM, int TN, int KT,
                                          int nw, int wv, int lane) {
  int wvu = __builtin_amdgcn_readfirstlane(wv);
  for (int t = wvu; t < TM * TN; t += nw) {
    int mt = t / TN, nt = t - mt * TN;
    v8f acc = {};
    for (int kt = 0; kt < KT; ++kt) {
      v16h a = load_a_frag(As, lda, mt, kt, lane);
      v16h b = Bsw[(size_t)(kt * TN + nt) * 32 + lane];
      acc = __builtin_amdgcn_wmma_f32_16x16x32_f16(false, a, false, b,
                                                   (short)0, acc, false, false);
    }
    int col = nt * 16 + (lane & 15);
    float bv = bias[col];
    int rbase = (lane >> 4) << 3;
#pragma unroll
    for (int r = 0; r < 8; ++r) {
      float v = acc[r] + bv;
      v = v > 0.0f ? v : 0.0f;
      Cs[(mt * 16 + rbase + r) * ldc + col] = (_Float16)v;
    }
  }
}

// Final layer: relu(A @ W + bias) followed by column-max over the 16-row tile.
// Writes per-(M-tile, column) partial maxes into maxp[(mtOff+mt)*N + col].
// ReLU => values >= 0, so a 0-initialized max equals the true max.
__device__ __forceinline__ void gemm_final(const _Float16* As, int lda,
                                           const v16h* __restrict__ Bsw,
                                           const float* __restrict__ bias,
                                           float* maxp, int N,
                                           int TM, int TN, int KT,
                                           int nw, int wv, int lane, int mtOff) {
  int wvu = __builtin_amdgcn_readfirstlane(wv);
  for (int t = wvu; t < TM * TN; t += nw) {
    int mt = t / TN, nt = t - mt * TN;
    v8f acc = {};
    for (int kt = 0; kt < KT; ++kt) {
      v16h a = load_a_frag(As, lda, mt, kt, lane);
      v16h b = Bsw[(size_t)(kt * TN + nt) * 32 + lane];
      acc = __builtin_amdgcn_wmma_f32_16x16x32_f16(false, a, false, b,
                                                   (short)0, acc, false, false);
    }
    int col = nt * 16 + (lane & 15);
    float bv = bias[col];
    float m = 0.0f;
#pragma unroll
    for (int r = 0; r < 8; ++r) {
      float v = acc[r] + bv;
      v = v > 0.0f ? v : 0.0f;
      m = fmaxf(m, v);
    }
    m = fmaxf(m, __shfl_xor(m, 16));   // combine rows 0-7 with 8-15 (same col)
    maxp[(mtOff + mt) * N + col] = m;  // both half-waves store identical value
  }
}

// ---------------------------------------------------------------------------
// FPS over 8192 points, 128 selections per batch. One block per batch.
// dist kept in registers; argmax via shfl + LDS cross-wave reduce.
// Tie-break: smallest index (matches jnp.argmax first-max semantics).
// ---------------------------------------------------------------------------
__global__ void fps1_kernel(const float* __restrict__ xyz, float* __restrict__ new_xyz) {
  const int b = blockIdx.x, tid = threadIdx.x;
  const float* X = xyz + (size_t)b * 8192 * 3;
  float d[32];
#pragma unroll
  for (int j = 0; j < 32; ++j) d[j] = 1e10f;
  __shared__ int   s_sel[128];
  __shared__ float s_wv[8];
  __shared__ int   s_wi[8];
  __shared__ int   s_last;
  if (tid == 0) { s_sel[0] = 0; s_last = 0; }
  __syncthreads();
  const int lane = tid & 31, wv = tid >> 5;
  for (int it = 1; it < 128; ++it) {
    int last = s_last;
    float px = X[last * 3], py = X[last * 3 + 1], pz = X[last * 3 + 2];
    float bv = -1.0f; int bi = 0;
#pragma unroll 4
    for (int j = 0; j < 32; ++j) {
      int i = tid + j * 256;
      float dx = X[i * 3] - px, dy = X[i * 3 + 1] - py, dz = X[i * 3 + 2] - pz;
      float dd = dx * dx + dy * dy + dz * dz;
      float nd = d[j] < dd ? d[j] : dd;
      d[j] = nd;
      if (nd > bv) { bv = nd; bi = i; }
    }
    for (int off = 16; off > 0; off >>= 1) {
      float ov = __shfl_down(bv, off);
      int   oi = __shfl_down(bi, off);
      if (ov > bv || (ov == bv && oi < bi)) { bv = ov; bi = oi; }
    }
    if (lane == 0) { s_wv[wv] = bv; s_wi[wv] = bi; }
    __syncthreads();
    if (tid == 0) {
      float fv = s_wv[0]; int fi = s_wi[0];
      for (int w = 1; w < 8; ++w) {
        float ov = s_wv[w]; int oi = s_wi[w];
        if (ov > fv || (ov == fv && oi < fi)) { fv = ov; fi = oi; }
      }
      s_sel[it] = fi; s_last = fi;
    }
    __syncthreads();
  }
  for (int t = tid; t < 128; t += 256) {
    int idx = s_sel[t];
    new_xyz[((size_t)b * 128 + t) * 3 + 0] = X[idx * 3 + 0];
    new_xyz[((size_t)b * 128 + t) * 3 + 1] = X[idx * 3 + 1];
    new_xyz[((size_t)b * 128 + t) * 3 + 2] = X[idx * 3 + 2];
  }
}

// FPS over 128 points, 32 selections. One wave per batch.
__global__ void fps2_kernel(const float* __restrict__ new_xyz, float* __restrict__ xyz2) {
  const int b = blockIdx.x, lane = threadIdx.x;
  const float* X = new_xyz + (size_t)b * 128 * 3;
  float d[4];
#pragma unroll
  for (int j = 0; j < 4; ++j) d[j] = 1e10f;
  int last = 0;
  if (lane == 0) {
    xyz2[(size_t)b * 32 * 3 + 0] = X[0];
    xyz2[(size_t)b * 32 * 3 + 1] = X[1];
    xyz2[(size_t)b * 32 * 3 + 2] = X[2];
  }
  for (int it = 1; it < 32; ++it) {
    float px = X[last * 3], py = X[last * 3 + 1], pz = X[last * 3 + 2];
    float bv = -1.0f; int bi = 0;
#pragma unroll
    for (int j = 0; j < 4; ++j) {
      int i = lane + j * 32;
      float dx = X[i * 3] - px, dy = X[i * 3 + 1] - py, dz = X[i * 3 + 2] - pz;
      float dd = dx * dx + dy * dy + dz * dz;
      float nd = d[j] < dd ? d[j] : dd;
      d[j] = nd;
      if (nd > bv) { bv = nd; bi = i; }
    }
    for (int off = 16; off > 0; off >>= 1) {
      float ov = __shfl_down(bv, off);
      int   oi = __shfl_down(bi, off);
      if (ov > bv || (ov == bv && oi < bi)) { bv = ov; bi = oi; }
    }
    bi = __shfl(bi, 0);
    last = bi;
    if (lane == 0) {
      xyz2[((size_t)b * 32 + it) * 3 + 0] = X[bi * 3 + 0];
      xyz2[((size_t)b * 32 + it) * 3 + 1] = X[bi * 3 + 1];
      xyz2[((size_t)b * 32 + it) * 3 + 2] = X[bi * 3 + 2];
    }
  }
}

// ---------------------------------------------------------------------------
// Ball query: first `nsamp` indices (ascending) with d2 < r2; pad with first.
// One wave per center; ordered compaction via ballot/popc.
// ---------------------------------------------------------------------------
__device__ __forceinline__ void ball_query_wave(const float* X, int n,
                                                float cx, float cy, float cz,
                                                float r2, int nsamp,
                                                int* __restrict__ out, int lane) {
  int count = 0, firstIdx = 0;
  const int chunks = n >> 5;
  for (int ch = 0; ch < chunks; ++ch) {
    int p = ch * 32 + lane;
    float dx = X[p * 3] - cx, dy = X[p * 3 + 1] - cy, dz = X[p * 3 + 2] - cz;
    float dd = dx * dx + dy * dy + dz * dz;
    bool within = dd < r2;
    unsigned m = (unsigned)__ballot(within);
    if (count == 0 && m) firstIdx = ch * 32 + (__ffs(m) - 1);
    int pos = count + __popc(m & ((1u << lane) - 1u));
    if (within && pos < nsamp) out[pos] = p;
    count += __popc(m);
    if (count >= nsamp) break;
  }
  if (count < nsamp) {
    for (int s = count + lane; s < nsamp; s += 32) out[s] = firstIdx;
  }
}

__global__ void ballq1_kernel(const float* __restrict__ xyz,
                              const float* __restrict__ new_xyz,
                              int* __restrict__ g_idx) {
  int lane = threadIdx.x & 31, wip = threadIdx.x >> 5;
  int g = blockIdx.x * 8 + wip;                 // 0..8191 (b*128+c)
  const float* X = xyz + (size_t)(g >> 7) * 8192 * 3;
  float cx = new_xyz[(size_t)g * 3], cy = new_xyz[(size_t)g * 3 + 1], cz = new_xyz[(size_t)g * 3 + 2];
  ball_query_wave(X, 8192, cx, cy, cz, 0.02f * 0.02f, 64, g_idx + (size_t)g * 64, lane);
}

__global__ void ballq2_kernel(const float* __restrict__ new_xyz,
                              const float* __restrict__ xyz2,
                              int* __restrict__ g_idx) {
  int lane = threadIdx.x & 31, wip = threadIdx.x >> 5;
  int g = blockIdx.x * 8 + wip;                 // 0..2047 (b*32+c)
  const float* X = new_xyz + (size_t)(g >> 5) * 128 * 3;
  float cx = xyz2[(size_t)g * 3], cy = xyz2[(size_t)g * 3 + 1], cz = xyz2[(size_t)g * 3 + 2];
  ball_query_wave(X, 128, cx, cy, cz, 0.04f * 0.04f, 128, g_idx + (size_t)g * 128, lane);
}

// ---------------------------------------------------------------------------
// SA1 grouped MLP: per (b, center): [64 x 3(pad32)] -> 64 -> 64 -> 128, max.
// Block = 128 threads (4 waves), one group per block.
// ---------------------------------------------------------------------------
__global__ void sa1_mlp_kernel(const float* __restrict__ xyz,
                               const float* __restrict__ new_xyz,
                               const int* __restrict__ g_idx,
                               const v16h* __restrict__ w1, const float* __restrict__ b1,
                               const v16h* __restrict__ w2, const float* __restrict__ b2,
                               const v16h* __restrict__ w3, const float* __restrict__ b3,
                               _Float16* __restrict__ f1h) {
  const int g = blockIdx.x;                     // b*128 + c
  const int tid = threadIdx.x, lane = tid & 31, wv = tid >> 5;
  __shared__ __align__(32) _Float16 bufA[64 * 64];   // A0 (64x32) / act2 (64x64)
  __shared__ __align__(32) _Float16 bufB[64 * 64];   // act1 (64x64)
  __shared__ float maxp[4 * 128];
  const float* X = xyz + (size_t)(g >> 7) * 8192 * 3;
  const float cx = new_xyz[(size_t)g * 3], cy = new_xyz[(size_t)g * 3 + 1], cz = new_xyz[(size_t)g * 3 + 2];
  if (tid < 64) {
    int idx = g_idx[(size_t)g * 64 + tid];
    _Float16* row = bufA + tid * 32;
    row[0] = (_Float16)(X[idx * 3 + 0] - cx);
    row[1] = (_Float16)(X[idx * 3 + 1] - cy);
    row[2] = (_Float16)(X[idx * 3 + 2] - cz);
#pragma unroll
    for (int k = 3; k < 32; ++k) row[k] = (_Float16)0.0f;
  }
  __syncthreads();
  gemm_relu(bufA, 32, w1, b1, bufB, 64, 4, 4, 1, 4, wv, lane);
  __syncthreads();
  gemm_relu(bufB, 64, w2, b2, bufA, 64, 4, 4, 2, 4, wv, lane);
  __syncthreads();
  gemm_final(bufA, 64, w3, b3, maxp, 128, 4, 8, 2, 4, wv, lane, 0);
  __syncthreads();
  for (int col = tid; col < 128; col += 128) {
    float m = maxp[col];
#pragma unroll
    for (int mt = 1; mt < 4; ++mt) m = fmaxf(m, maxp[mt * 128 + col]);
    f1h[(size_t)g * 128 + col] = (_Float16)m;
  }
}

// ---------------------------------------------------------------------------
// SA2 grouped MLP: per (b, center): [128 x 131(pad160)] -> 128 -> 128 -> 256, max.
// A0 K-layout is feats[0..127], xyz[128..130], pad[131..159] (weights permuted
// to match in the swizzle), so the feature block copies as aligned uint4.
// 128 sample rows processed in two 64-row passes to bound LDS. Block = 256 thr.
// ---------------------------------------------------------------------------
__global__ void sa2_mlp_kernel(const float* __restrict__ new_xyz,
                               const float* __restrict__ xyz2,
                               const int* __restrict__ g_idx,
                               const _Float16* __restrict__ f1h,
                               const v16h* __restrict__ w1, const float* __restrict__ b1,
                               const v16h* __restrict__ w2, const float* __restrict__ b2,
                               const v16h* __restrict__ w3, const float* __restrict__ b3,
                               _Float16* __restrict__ f2h) {
  const int g = blockIdx.x;                     // b*32 + c
  const int b = g >> 5;
  const int tid = threadIdx.x, lane = tid & 31, wv = tid >> 5;
  __shared__ __align__(32) _Float16 bufA[64 * 160];  // A0 (64x160) / act2 (64x128)
  __shared__ __align__(32) _Float16 bufB[64 * 128];  // act1 (64x128)
  __shared__ float maxp[8 * 256];
  const float* P = new_xyz + (size_t)b * 128 * 3;
  const float cx = xyz2[(size_t)g * 3], cy = xyz2[(size_t)g * 3 + 1], cz = xyz2[(size_t)g * 3 + 2];
  for (int pass = 0; pass < 2; ++pass) {
    if (tid < 64) {
      int s = pass * 64 + tid;
      int idx = g_idx[(size_t)g * 128 + s];
      _Float16* row = bufA + tid * 160;          // 320B stride, 16B-aligned
      const uint4* fsrc = (const uint4*)(f1h + ((size_t)b * 128 + idx) * 128);
      uint4* fdst = (uint4*)row;
#pragma unroll
      for (int j = 0; j < 16; ++j) fdst[j] = fsrc[j];   // feats: 128 halves
      row[128] = (_Float16)(P[idx * 3 + 0] - cx);
      row[129] = (_Float16)(P[idx * 3 + 1] - cy);
      row[130] = (_Float16)(P[idx * 3 + 2] - cz);
#pragma unroll
      for (int k = 131; k < 160; ++k) row[k] = (_Float16)0.0f;
    }
    __syncthreads();
    gemm_relu(bufA, 160, w1, b1, bufB, 128, 4, 8, 5, 8, wv, lane);
    __syncthreads();
    gemm_relu(bufB, 128, w2, b2, bufA, 128, 4, 8, 4, 8, wv, lane);
    __syncthreads();
    gemm_final(bufA, 128, w3, b3, maxp, 256, 4, 16, 4, 8, wv, lane, pass * 4);
    __syncthreads();
  }
  for (int col = tid; col < 256; col += 256) {
    float m = maxp[col];
#pragma unroll
    for (int mt = 1; mt < 8; ++mt) m = fmaxf(m, maxp[mt * 256 + col]);
    f2h[(size_t)g * 256 + col] = (_Float16)m;
  }
}

// ---------------------------------------------------------------------------
// SA3: per batch: [32 x 259(pad288)] -> 256 -> 256 -> 512, max over 32 -> f3.
// A0 K-layout: f2[0..255], xyz2[256..258], pad[259..287] (weights permuted).
// Block = 256 threads, one batch per block.
// ---------------------------------------------------------------------------
__global__ void sa3_mlp_kernel(const float* __restrict__ xyz2,
                               const _Float16* __restrict__ f2h,
                               const v16h* __restrict__ w1, const float* __restrict__ b1,
                               const v16h* __restrict__ w2, const float* __restrict__ b2,
                               const v16h* __restrict__ w3, const float* __restrict__ b3,
                               float* __restrict__ f3) {
  const int b = blockIdx.x;
  const int tid = threadIdx.x, lane = tid & 31, wv = tid >> 5;
  __shared__ __align__(32) _Float16 bufA[32 * 288];  // A0 (32x288) / act2 (32x256)
  __shared__ __align__(32) _Float16 bufB[32 * 256];  // act1 (32x256)
  __shared__ float maxp[2 * 512];
  if (tid < 32) {
    _Float16* row = bufA + tid * 288;            // 576B stride, 16B-aligned
    const uint4* fsrc = (const uint4*)(f2h + ((size_t)b * 32 + tid) * 256);
    uint4* fdst = (uint4*)row;
#pragma unroll
    for (int j = 0; j < 32; ++j) fdst[j] = fsrc[j];   // f2: 256 halves
    row[256] = (_Float16)xyz2[((size_t)b * 32 + tid) * 3 + 0];
    row[257] = (_Float16)xyz2[((size_t)b * 32 + tid) * 3 + 1];
    row[258] = (_Float16)xyz2[((size_t)b * 32 + tid) * 3 + 2];
#pragma unroll
    for (int k = 259; k < 288; ++k) row[k] = (_Float16)0.0f;
  }
  __syncthreads();
  gemm_relu(bufA, 288, w1, b1, bufB, 256, 2, 16, 9, 8, wv, lane);
  __syncthreads();
  gemm_relu(bufB, 256, w2, b2, bufA, 256, 2, 16, 8, 8, wv, lane);
  __syncthreads();
  gemm_final(bufA, 256, w3, b3, maxp, 512, 2, 32, 8, 8, wv, lane, 0);
  __syncthreads();
  for (int col = tid; col < 512; col += 256) {
    float m = fmaxf(maxp[col], maxp[512 + col]);
    f3[(size_t)b * 512 + col] = m;
  }
}

// FC head: out = relu(f3 @ W1 + b1) @ W2 + b2.  One block per batch.
__global__ void fc_kernel(const float* __restrict__ f3,
                          const float* __restrict__ w1, const float* __restrict__ b1,
                          const float* __restrict__ w2, const float* __restrict__ b2,
                          float* __restrict__ out) {
  const int b = blockIdx.x, t = threadIdx.x;
  __shared__ float fs[512];
  __shared__ float hs[256];
  for (int i = t; i < 512; i += 256) fs[i] = f3[(size_t)b * 512 + i];
  __syncthreads();
  {
    float s = b1[t];
    for (int k = 0; k < 512; ++k) s += fs[k] * w1[(size_t)k * 256 + t];
    hs[t] = s > 0.0f ? s : 0.0f;
  }
  __syncthreads();
  if (t < 6) {
    float s = b2[t];
    for (int k = 0; k < 256; ++k) s += hs[k] * w2[(size_t)k * 6 + t];
    out[(size_t)b * 6 + t] = s;
  }
}

// ---------------------------------------------------------------------------
extern "C" void kernel_launch(void* const* d_in, const int* in_sizes, int n_in,
                              void* d_out, int out_size, void* d_ws, size_t ws_size,
                              hipStream_t stream) {
  const float* xyz = (const float*)d_in[0];
  // sa1: w1,b1,w2,b2,w3,b3 = d_in[1..6]; sa2: 7..12; sa3: 13..18; fc: 19..22
  const float* s1w1 = (const float*)d_in[1];  const float* s1b1 = (const float*)d_in[2];
  const float* s1w2 = (const float*)d_in[3];  const float* s1b2 = (const float*)d_in[4];
  const float* s1w3 = (const float*)d_in[5];  const float* s1b3 = (const float*)d_in[6];
  const float* s2w1 = (const float*)d_in[7];  const float* s2b1 = (const float*)d_in[8];
  const float* s2w2 = (const float*)d_in[9];  const float* s2b2 = (const float*)d_in[10];
  const float* s2w3 = (const float*)d_in[11]; const float* s2b3 = (const float*)d_in[12];
  const float* s3w1 = (const float*)d_in[13]; const float* s3b1 = (const float*)d_in[14];
  const float* s3w2 = (const float*)d_in[15]; const float* s3b2 = (const float*)d_in[16];
  const float* s3w3 = (const float*)d_in[17]; const float* s3b3 = (const float*)d_in[18];
  const float* fw1  = (const float*)d_in[19]; const float* fb1  = (const float*)d_in[20];
  const float* fw2  = (const float*)d_in[21]; const float* fb2  = (const float*)d_in[22];

  char* ws = (char*)d_ws;
  size_t o = 0;
  auto take = [&](size_t bytes) -> char* {
    char* p = ws + o;
    o += (bytes + 255) & ~(size_t)255;
    return p;
  };
  float*     new_xyz = (float*)take((size_t)64 * 128 * 3 * 4);
  float*     xyz2    = (float*)take((size_t)64 * 32 * 3 * 4);
  int*       gidx1   = (int*)take((size_t)64 * 128 * 64 * 4);
  int*       gidx2   = (int*)take((size_t)64 * 32 * 128 * 4);
  _Float16*  f1h     = (_Float16*)take((size_t)64 * 128 * 128 * 2);
  _Float16*  f2h     = (_Float16*)take((size_t)64 * 32 * 256 * 2);
  float*     f3      = (float*)take((size_t)64 * 512 * 4);
  _Float16*  w1s1 = (_Float16*)take((size_t)32 * 64 * 2);
  _Float16*  w2s1 = (_Float16*)take((size_t)64 * 64 * 2);
  _Float16*  w3s1 = (_Float16*)take((size_t)64 * 128 * 2);
  _Float16*  w1s2 = (_Float16*)take((size_t)160 * 128 * 2);
  _Float16*  w2s2 = (_Float16*)take((size_t)128 * 128 * 2);
  _Float16*  w3s2 = (_Float16*)take((size_t)128 * 256 * 2);
  _Float16*  w1s3 = (_Float16*)take((size_t)288 * 256 * 2);
  _Float16*  w2s3 = (_Float16*)take((size_t)256 * 256 * 2);
  _Float16*  w3s3 = (_Float16*)take((size_t)256 * 512 * 2);

  auto swz = [&](const float* src, _Float16* dst, int Kin, int Kpad, int N, int feat3) {
    int total = Kpad * N;
    swz_kernel<<<(total + 255) / 256, 256, 0, stream>>>(src, dst, Kin, Kpad, N, feat3);
  };
  swz(s1w1, w1s1, 3, 32, 64, 0);
  swz(s1w2, w2s1, 64, 64, 64, 0);
  swz(s1w3, w3s1, 64, 64, 128, 0);
  swz(s2w1, w1s2, 131, 160, 128, 128);   // feats-first K permutation
  swz(s2w2, w2s2, 128, 128, 128, 0);
  swz(s2w3, w3s2, 128, 128, 256, 0);
  swz(s3w1, w1s3, 259, 288, 256, 256);   // feats-first K permutation
  swz(s3w2, w2s3, 256, 256, 256, 0);
  swz(s3w3, w3s3, 256, 256, 512, 0);

  fps1_kernel<<<64, 256, 0, stream>>>(xyz, new_xyz);
  ballq1_kernel<<<1024, 256, 0, stream>>>(xyz, new_xyz, gidx1);
  sa1_mlp_kernel<<<8192, 128, 0, stream>>>(xyz, new_xyz, gidx1,
      (const v16h*)w1s1, s1b1, (const v16h*)w2s1, s1b2, (const v16h*)w3s1, s1b3, f1h);
  fps2_kernel<<<64, 32, 0, stream>>>(new_xyz, xyz2);
  ballq2_kernel<<<256, 256, 0, stream>>>(new_xyz, xyz2, gidx2);
  sa2_mlp_kernel<<<2048, 256, 0, stream>>>(new_xyz, xyz2, gidx2, f1h,
      (const v16h*)w1s2, s2b1, (const v16h*)w2s2, s2b2, (const v16h*)w3s2, s2b3, f2h);
  sa3_mlp_kernel<<<64, 256, 0, stream>>>(xyz2, f2h,
      (const v16h*)w1s3, s3b1, (const v16h*)w2s3, s3b2, (const v16h*)w3s3, s3b3, f3);
  fc_kernel<<<64, 256, 0, stream>>>(f3, fw1, fb1, fw2, fb2, (float*)d_out);
}
